// CrossOscillatorAttention_22393959481777
// MI455X (gfx1250) — matrix operand order repaired and checked by reading, not compiled, not run on hardware
//
#include <hip/hip_runtime.h>

// ============================================================================
// CrossOscillatorAttention for MI455X (gfx1250, wave32, WMMA bf16 16x16x32)
//
// ws layout (floats, P = 4*128*4096 = 2097152):
//   [0]      s (pooled src)        [1P] r (pooled ref)
//   per evolve e in {q,k,v}: base = (2+7e)P :
//        force, omega, zeta, x_a, x_b, v, pw_tmp
//   [23P] ctx_pool  [24P] ctx_out  [25P] norms (4*16384)
//   bf16 weights after that (~512KB).  Total ~211 MB of workspace.
// ============================================================================

typedef __attribute__((ext_vector_type(16))) __bf16 v16bf;
typedef __attribute__((ext_vector_type(8)))  __bf16 v8bf;
typedef __attribute__((ext_vector_type(8)))  float  v8f;

#define WMMA_BF16(A, Bm, C) \
  __builtin_amdgcn_wmma_f32_16x16x32_bf16(false, (A), false, (Bm), (short)0, (C), false, false)

__device__ __forceinline__ __bf16 f2bf(float f) {
  unsigned u = __builtin_bit_cast(unsigned, f);
  unsigned short h = (unsigned short)((u + 0x7fffu + ((u >> 16) & 1u)) >> 16);
  return __builtin_bit_cast(__bf16, h);
}
// pack two floats into one dword of bf16 (RNE), for b32 LDS staging stores
__device__ __forceinline__ unsigned pkbf(float lo, float hi) {
  unsigned a = (unsigned)__builtin_bit_cast(unsigned short, f2bf(lo));
  unsigned b = (unsigned)__builtin_bit_cast(unsigned short, f2bf(hi));
  return a | (b << 16);
}

__device__ __forceinline__ float sigm(float v) { return 1.f / (1.f + __expf(-v)); }

// A fragment (16 rows x 32 K) from a [rows][ld] bf16 tile. Per ISA 7.12.2:
// lanes 0-15 hold K {0..7,16..23}, lanes 16-31 hold K {8..15,24..31}.
__device__ __forceinline__ v16bf ld_afrag(const __bf16* tb, int ld, int lane, int rowb, int kb) {
  int row = rowb + (lane & 15);
  int ko  = kb * 32 + ((lane >> 4) & 1) * 8;
  const __bf16* p = tb + row * ld + ko;
  v8bf lo = *(const v8bf*)p;
  v8bf hi = *(const v8bf*)(p + 16);
  return __builtin_shufflevector(lo, hi, 0,1,2,3,4,5,6,7,8,9,10,11,12,13,14,15);
}

// B fragment (32 K x 16 cols) from a [col][ld] bf16 tile (column-major-by-col):
// lane n<16 -> col n, K {0..7,16..23}; lane>=16 -> col n-16, K {8..15,24..31}.
__device__ __forceinline__ v16bf ld_bfrag(const __bf16* tb, int ld, int lane, int colb, int kb) {
  int col = colb + (lane & 15);
  int ko  = kb * 32 + ((lane >> 4) & 1) * 8;
  const __bf16* p = tb + col * ld + ko;
  v8bf lo = *(const v8bf*)p;
  v8bf hi = *(const v8bf*)(p + 16);
  return __builtin_shufflevector(lo, hi, 0,1,2,3,4,5,6,7,8,9,10,11,12,13,14,15);
}

// ---------------------------------------------------------------------------
__global__ void k_cvt(const float* __restrict__ s, __bf16* __restrict__ d, int n) {
  int i = blockIdx.x * 256 + threadIdx.x;
  if (i < n) d[i] = f2bf(s[i]);
}

// ---------------------------------------------------------------------------
__global__ __launch_bounds__(256) void k_avgpool(const float* __restrict__ in,
                                                 float* __restrict__ out) {
  int plane = blockIdx.x;                       // b*128 + c
  const float* ip = in + (size_t)plane * 65536;
  float* op = out + (size_t)plane * 4096;
  for (int it = 0; it < 16; ++it) {
    int p = it * 256 + threadIdx.x;
    int oy = p >> 6, ox = p & 63;
    float s = 0.f;
#pragma unroll
    for (int dy = 0; dy < 4; ++dy) {
      float4 v = *(const float4*)(ip + (oy * 4 + dy) * 256 + ox * 4);
      s += v.x + v.y + v.z + v.w;
    }
    op[p] = s * 0.0625f;
  }
}

// ---------------------------------------------------------------------------
// conv1x1 over pooled (B,128,4096) planar tensors. out[o][p] = act(sum_c W[o][c] in[c][p]).
// MODE 0: plain; 1: 2*sigmoid; 2: sigmoid; 3: plain + xout=f*DT^2, vout=f*DT (osc. step 1)
template <int MODE>
__global__ __launch_bounds__(256) void k_conv1x1(const float* __restrict__ in,
                                                 const __bf16* __restrict__ Wbf,
                                                 float* __restrict__ out,
                                                 float* __restrict__ xout,
                                                 float* __restrict__ vout) {
  const int HW = 4096;
  __shared__ __align__(16) __bf16 As[64 * 136];
  int b = blockIdx.y;
  int pixb = blockIdx.x * 64;
  const float* inb = in + (size_t)b * 128 * HW;
  int t = threadIdx.x;
  {  // transpose-stage planar -> [pix][ch], packed b32 stores
    int pix = t & 63, c0 = (t >> 6) * 2;
    for (int c = c0; c < 128; c += 8) {
      float lo = inb[(size_t)c * HW + pixb + pix];
      float hi = inb[(size_t)(c + 1) * HW + pixb + pix];
      *(unsigned*)&As[pix * 136 + c] = pkbf(lo, hi);
    }
  }
  __syncthreads();
  int wv = t >> 5, lane = t & 31;
  int r0 = ((lane >> 4) & 1) * 8;
  for (int rt = 0; rt < 4; ++rt) {
    v8f acc = {0.f, 0.f, 0.f, 0.f, 0.f, 0.f, 0.f, 0.f};
#pragma unroll
    for (int kb = 0; kb < 4; ++kb) {
      v16bf a  = ld_afrag(As, 136, lane, rt * 16, kb);
      v16bf bm = ld_bfrag(Wbf, 128, lane, wv * 16, kb);
      acc = WMMA_BF16(a, bm, acc);
    }
    int ch = wv * 16 + (lane & 15);
    int rowb = pixb + rt * 16 + r0;
    float fv[8];
#pragma unroll
    for (int i = 0; i < 8; ++i) {
      float v = acc[i];
      if constexpr (MODE == 1) v = 2.f * sigm(v);
      if constexpr (MODE == 2) v = sigm(v);
      fv[i] = v;
    }
    float* op = out + (size_t)(b * 128 + ch) * HW + rowb;
    *(float4*)op       = make_float4(fv[0], fv[1], fv[2], fv[3]);
    *(float4*)(op + 4) = make_float4(fv[4], fv[5], fv[6], fv[7]);
    if constexpr (MODE == 3) {
      float* xp = xout + (size_t)(b * 128 + ch) * HW + rowb;
      float* vp = vout + (size_t)(b * 128 + ch) * HW + rowb;
      *(float4*)xp       = make_float4(fv[0]*0.04f, fv[1]*0.04f, fv[2]*0.04f, fv[3]*0.04f);
      *(float4*)(xp + 4) = make_float4(fv[4]*0.04f, fv[5]*0.04f, fv[6]*0.04f, fv[7]*0.04f);
      *(float4*)vp       = make_float4(fv[0]*0.2f, fv[1]*0.2f, fv[2]*0.2f, fv[3]*0.2f);
      *(float4*)(vp + 4) = make_float4(fv[4]*0.2f, fv[5]*0.2f, fv[6]*0.2f, fv[7]*0.2f);
    }
  }
}

// ---------------------------------------------------------------------------
// depthwise 3x3 (zero pad) + oscillator integration; reads x_in, writes x_out, v in place.
__global__ __launch_bounds__(256) void k_dwstep(const float* __restrict__ xin,
                                                const float* __restrict__ pw,
                                                const float* __restrict__ fo,
                                                const float* __restrict__ om,
                                                const float* __restrict__ ze,
                                                float* __restrict__ vv,
                                                float* __restrict__ xout,
                                                const float* __restrict__ Wdw) {
  const float DT = 0.2f;
  int plane = blockIdx.x;                         // b*128 + c
  int c = plane & 127;
  size_t po = (size_t)plane * 4096;
  const float* xp = xin + po;
  float w[9];
#pragma unroll
  for (int i = 0; i < 9; ++i) w[i] = Wdw[c * 9 + i];
  for (int it = 0; it < 16; ++it) {
    int p = it * 256 + threadIdx.x;
    int y = p >> 6, x = p & 63;
    float s = 0.f;
#pragma unroll
    for (int dy = -1; dy <= 1; ++dy) {
      int yy = y + dy;
      if (yy < 0 || yy > 63) continue;
#pragma unroll
      for (int dx = -1; dx <= 1; ++dx) {
        int xx = x + dx;
        if (xx < 0 || xx > 63) continue;
        s += w[(dy + 1) * 3 + (dx + 1)] * xp[yy * 64 + xx];
      }
    }
    float cc = s + pw[po + p];
    float o = om[po + p], z = ze[po + p], f = fo[po + p];
    float vo = vv[po + p], xc = xp[p];
    float a = f + cc - 2.f * z * o * vo - o * o * xc;
    vo += a * DT;
    vv[po + p] = vo;
    xout[po + p] = xc + vo * DT;
  }
}

// ---------------------------------------------------------------------------
__global__ __launch_bounds__(256) void k_norm(const float* __restrict__ x,
                                              float* __restrict__ n) {
  int b = blockIdx.x >> 4, ch = blockIdx.x & 15;
  int p = ch * 256 + threadIdx.x;
  const float* xp = x + (size_t)b * 128 * 4096 + p;
  float a = 0.f;
  for (int c = 0; c < 128; ++c) { float v = xp[(size_t)c * 4096]; a += v * v; }
  n[b * 4096 + p] = a;
}

// ---------------------------------------------------------------------------
// Flash attention, 64 queries x 64 keys per tile, logits = qk/sqrt(C)
//   - 0.15*(||qw-kw|| + ||qz-kz||)/sqrt(C), online softmax, O += P@V.
__global__ __launch_bounds__(128) void k_flash(const float* __restrict__ q,
                                               const float* __restrict__ kq,
                                               const float* __restrict__ vv,
                                               const float* __restrict__ qw,
                                               const float* __restrict__ kw,
                                               const float* __restrict__ qz,
                                               const float* __restrict__ kz,
                                               const float* __restrict__ nqw,
                                               const float* __restrict__ nqz,
                                               const float* __restrict__ nkw,
                                               const float* __restrict__ nkz,
                                               float* __restrict__ ctx) {
  const int HW = 4096;
  __shared__ __align__(16) __bf16 sm[3 * 64 * 136 + 128 * 72 + 4 * 16 * 72];
  __bf16* ks  = sm;                     // reused for q-staging at start
  __bf16* kws = sm + 64 * 136;
  __bf16* kzs = sm + 2 * 64 * 136;
  __bf16* vs  = sm + 3 * 64 * 136;      // [128 ch][72]
  __bf16* Ps  = vs + 128 * 72;          // [4 waves][16][72]

  int b = blockIdx.y, qb = blockIdx.x * 64;
  size_t bo = (size_t)b * 128 * HW;
  int t = threadIdx.x, lane = t & 31, wv = t >> 5;
  int r0 = ((lane >> 4) & 1) * 8;

  {  // stage q/qw/qz (transpose planar -> [pix][ch], packed b32 stores)
    int pix = t & 63, c0 = (t >> 6) * 2;
    for (int c = c0; c < 128; c += 4) {
      size_t g0 = bo + (size_t)c * HW + qb + pix;
      *(unsigned*)&ks [pix * 136 + c] = pkbf(q [g0], q [g0 + HW]);
      *(unsigned*)&kws[pix * 136 + c] = pkbf(qw[g0], qw[g0 + HW]);
      *(unsigned*)&kzs[pix * 136 + c] = pkbf(qz[g0], qz[g0 + HW]);
    }
  }
  __syncthreads();
  v16bf fq[4], fw[4], fz[4];
#pragma unroll
  for (int kb = 0; kb < 4; ++kb) {
    fq[kb] = ld_afrag(ks,  136, lane, wv * 16, kb);
    fw[kb] = ld_afrag(kws, 136, lane, wv * 16, kb);
    fz[kb] = ld_afrag(kzs, 136, lane, wv * 16, kb);
  }
  float nqw_r[8], nqz_r[8];
  int rbase = qb + wv * 16 + r0;
#pragma unroll
  for (int i = 0; i < 8; ++i) {
    nqw_r[i] = nqw[b * 4096 + rbase + i];
    nqz_r[i] = nqz[b * 4096 + rbase + i];
  }

  v8f O[8];
#pragma unroll
  for (int cb = 0; cb < 8; ++cb) O[cb] = (v8f){0.f,0.f,0.f,0.f,0.f,0.f,0.f,0.f};
  float m_[8], l_[8];
#pragma unroll
  for (int i = 0; i < 8; ++i) { m_[i] = -1e30f; l_[i] = 0.f; }
  const float scale = 0.08838834764831845f;   // 1/sqrt(128)

  for (int kt = 0; kt < 64; ++kt) {
    int kbse = kt * 64;
    __syncthreads();
    {  // stage K/KW/KZ (transposed) + V (direct), prefetch next tile
      int pix = t & 63, c0 = (t >> 6) * 2;
      for (int c = c0; c < 128; c += 4) {
        size_t g0 = bo + (size_t)c * HW + kbse + pix;
        *(unsigned*)&ks [pix * 136 + c] = pkbf(kq[g0], kq[g0 + HW]);
        *(unsigned*)&kws[pix * 136 + c] = pkbf(kw[g0], kw[g0 + HW]);
        *(unsigned*)&kzs[pix * 136 + c] = pkbf(kz[g0], kz[g0 + HW]);
      }
      const float* vp = vv + bo + (size_t)t * HW + kbse;
      for (int j = 0; j < 64; j += 4) {
        float4 v4 = *(const float4*)(vp + j);
        *(unsigned*)&vs[t * 72 + j]     = pkbf(v4.x, v4.y);
        *(unsigned*)&vs[t * 72 + j + 2] = pkbf(v4.z, v4.w);
      }
      if (kt + 1 < 64) {  // global_prefetch_b8 of next tile
        __builtin_prefetch(kq + bo + (size_t)t * HW + kbse + 64, 0, 1);
        __builtin_prefetch(vv + bo + (size_t)t * HW + kbse + 64, 0, 1);
      }
    }
    __syncthreads();

    float p_[4][8], mt[8];
#pragma unroll
    for (int i = 0; i < 8; ++i) mt[i] = -1e30f;
#pragma unroll
    for (int ct = 0; ct < 4; ++ct) {
      v8f d0 = {0.f,0.f,0.f,0.f,0.f,0.f,0.f,0.f};
      v8f d1 = d0, d2 = d0;
#pragma unroll
      for (int kb = 0; kb < 4; ++kb) {
        d0 = WMMA_BF16(fq[kb], ld_bfrag(ks,  136, lane, ct * 16, kb), d0);
        d1 = WMMA_BF16(fw[kb], ld_bfrag(kws, 136, lane, ct * 16, kb), d1);
        d2 = WMMA_BF16(fz[kb], ld_bfrag(kzs, 136, lane, ct * 16, kb), d2);
      }
      int mcol = kbse + ct * 16 + (lane & 15);
      float nw = nkw[b * 4096 + mcol], nz = nkz[b * 4096 + mcol];
#pragma unroll
      for (int i = 0; i < 8; ++i) {
        float dw = sqrtf(fmaxf(nqw_r[i] + nw - 2.f * d1[i], 1e-12f));
        float dz = sqrtf(fmaxf(nqz_r[i] + nz - 2.f * d2[i], 1e-12f));
        float s = d0[i] * scale - 0.15f * (dw + dz) * scale;
        p_[ct][i] = s;
        mt[i] = fmaxf(mt[i], s);
      }
    }
#pragma unroll
    for (int i = 0; i < 8; ++i)
      for (int msk = 1; msk < 16; msk <<= 1) mt[i] = fmaxf(mt[i], __shfl_xor(mt[i], msk, 32));
    float sc[8], lt[8];
#pragma unroll
    for (int i = 0; i < 8; ++i) {
      float mn = fmaxf(m_[i], mt[i]);
      sc[i] = __expf(m_[i] - mn);
      m_[i] = mn;
      float s = 0.f;
#pragma unroll
      for (int ct = 0; ct < 4; ++ct) { float e = __expf(p_[ct][i] - mn); p_[ct][i] = e; s += e; }
      lt[i] = s;
    }
#pragma unroll
    for (int i = 0; i < 8; ++i)
      for (int msk = 1; msk < 16; msk <<= 1) lt[i] += __shfl_xor(lt[i], msk, 32);
#pragma unroll
    for (int i = 0; i < 8; ++i) l_[i] = l_[i] * sc[i] + lt[i];

    {  // P (D-layout) -> LDS in A-layout geometry
      __bf16* Pw = Ps + wv * 16 * 72;
#pragma unroll
      for (int ct = 0; ct < 4; ++ct) {
        int col = ct * 16 + (lane & 15);
#pragma unroll
        for (int i = 0; i < 8; ++i) Pw[(r0 + i) * 72 + col] = f2bf(p_[ct][i]);
      }
    }
    __syncthreads();
    v16bf pa0 = ld_afrag(Ps + wv * 16 * 72, 72, lane, 0, 0);
    v16bf pa1 = ld_afrag(Ps + wv * 16 * 72, 72, lane, 0, 1);
#pragma unroll
    for (int cb = 0; cb < 8; ++cb) {
#pragma unroll
      for (int i = 0; i < 8; ++i) O[cb][i] *= sc[i];
      O[cb] = WMMA_BF16(pa0, ld_bfrag(vs, 72, lane, cb * 16, 0), O[cb]);
      O[cb] = WMMA_BF16(pa1, ld_bfrag(vs, 72, lane, cb * 16, 1), O[cb]);
    }
  }

  float inv[8];
#pragma unroll
  for (int i = 0; i < 8; ++i) inv[i] = 1.f / l_[i];
#pragma unroll
  for (int cb = 0; cb < 8; ++cb) {
    int ch = cb * 16 + (lane & 15);
    float* cp = ctx + bo + (size_t)ch * HW + qb + wv * 16 + r0;
    *(float4*)cp = make_float4(O[cb][0]*inv[0], O[cb][1]*inv[1], O[cb][2]*inv[2], O[cb][3]*inv[3]);
    *(float4*)(cp + 4) = make_float4(O[cb][4]*inv[4], O[cb][5]*inv[5], O[cb][6]*inv[6], O[cb][7]*inv[7]);
  }
}

// ---------------------------------------------------------------------------
// Fused: bilinear upsample ctx (pooled, L2-hot) + concat + Wg1(leaky) + Wg2(sigmoid)
// + residual. Single read of src, single write of out.
__global__ __launch_bounds__(256) void k_gating(const float* __restrict__ src,
                                                const float* __restrict__ ctxo,
                                                const __bf16* __restrict__ Wg1,
                                                const __bf16* __restrict__ Wg2,
                                                float* __restrict__ out) {
  const int HWF = 65536;
  __shared__ __align__(16) __bf16 Asc[64 * 264];   // [pix][src 0..127 | ctx 128..255]
  __shared__ __align__(16) __bf16 Hs[64 * 136];
  __shared__ float ctxf[64 * 132];
  int b = blockIdx.y, pixb = blockIdx.x * 64;
  int t = threadIdx.x;
  {
    int pix = t & 63;
    int gp = pixb + pix;
    const float* sb = src + (size_t)b * 128 * HWF + gp;
    {  // src half, packed b32 stores
      int c0 = (t >> 6) * 2;
      for (int c = c0; c < 128; c += 8)
        *(unsigned*)&Asc[pix * 264 + c] = pkbf(sb[(size_t)c * HWF], sb[(size_t)(c + 1) * HWF]);
    }
    // ctx half: bilinear sample from pooled ctx (half-pixel, edge clamp)
    int y = gp >> 8, x = gp & 255;
    float yf = (y + 0.5f) * 0.25f - 0.5f;
    float xf = (x + 0.5f) * 0.25f - 0.5f;
    float y0f = floorf(yf), x0f = floorf(xf);
    float fy = yf - y0f, fx = xf - x0f;
    int iy0 = min(max((int)y0f, 0), 63), iy1 = min(max((int)y0f + 1, 0), 63);
    int ix0 = min(max((int)x0f, 0), 63), ix1 = min(max((int)x0f + 1, 0), 63);
    const float* cbase = ctxo + (size_t)b * 128 * 4096;
    int c0 = (t >> 6) * 2;
    for (int c = c0; c < 128; c += 8) {
      float v2[2];
#pragma unroll
      for (int j = 0; j < 2; ++j) {
        const float* p = cbase + (size_t)(c + j) * 4096;
        float a00 = p[iy0 * 64 + ix0], a01 = p[iy0 * 64 + ix1];
        float a10 = p[iy1 * 64 + ix0], a11 = p[iy1 * 64 + ix1];
        v2[j] = (1.f - fy) * ((1.f - fx) * a00 + fx * a01) +
                fy * ((1.f - fx) * a10 + fx * a11);
        ctxf[pix * 132 + c + j] = v2[j];
      }
      *(unsigned*)&Asc[pix * 264 + 128 + c] = pkbf(v2[0], v2[1]);
    }
  }
  __syncthreads();
  int wv = t >> 5, lane = t & 31;
  int r0 = ((lane >> 4) & 1) * 8;
  for (int rt = 0; rt < 4; ++rt) {           // h = leaky(Wg1 @ [src;ctx])
    v8f acc = {0.f,0.f,0.f,0.f,0.f,0.f,0.f,0.f};
#pragma unroll
    for (int kb = 0; kb < 8; ++kb) {
      v16bf a  = ld_afrag(Asc, 264, lane, rt * 16, kb);
      v16bf bm = ld_bfrag(Wg1, 256, lane, wv * 16, kb);
      acc = WMMA_BF16(a, bm, acc);
    }
    int ch = wv * 16 + (lane & 15);
#pragma unroll
    for (int i = 0; i < 8; ++i) {
      float h = acc[i];
      h = h > 0.f ? h : 0.2f * h;
      Hs[(rt * 16 + r0 + i) * 136 + ch] = f2bf(h);
    }
  }
  __syncthreads();
  for (int rt = 0; rt < 4; ++rt) {           // g = sigmoid(Wg2 @ h); out = src + g*ctx
    v8f acc = {0.f,0.f,0.f,0.f,0.f,0.f,0.f,0.f};
#pragma unroll
    for (int kb = 0; kb < 4; ++kb) {
      v16bf a  = ld_afrag(Hs, 136, lane, rt * 16, kb);
      v16bf bm = ld_bfrag(Wg2, 128, lane, wv * 16, kb);
      acc = WMMA_BF16(a, bm, acc);
    }
    int ch = wv * 16 + (lane & 15);
    int rowb = rt * 16 + r0;
    const float* sp = src + (size_t)(b * 128 + ch) * HWF + pixb + rowb;
    float* op = out + (size_t)(b * 128 + ch) * HWF + pixb + rowb;
    float4 s0 = *(const float4*)sp, s1 = *(const float4*)(sp + 4);
    float4 o0, o1;
    o0.x = s0.x + sigm(acc[0]) * ctxf[(rowb + 0) * 132 + ch];
    o0.y = s0.y + sigm(acc[1]) * ctxf[(rowb + 1) * 132 + ch];
    o0.z = s0.z + sigm(acc[2]) * ctxf[(rowb + 2) * 132 + ch];
    o0.w = s0.w + sigm(acc[3]) * ctxf[(rowb + 3) * 132 + ch];
    o1.x = s1.x + sigm(acc[4]) * ctxf[(rowb + 4) * 132 + ch];
    o1.y = s1.y + sigm(acc[5]) * ctxf[(rowb + 5) * 132 + ch];
    o1.z = s1.z + sigm(acc[6]) * ctxf[(rowb + 6) * 132 + ch];
    o1.w = s1.w + sigm(acc[7]) * ctxf[(rowb + 7) * 132 + ch];
    *(float4*)op = o0;
    *(float4*)(op + 4) = o1;
  }
}

// ===========================================================================
extern "C" void kernel_launch(void* const* d_in, const int* in_sizes, int n_in,
                              void* d_out, int out_size, void* d_ws, size_t ws_size,
                              hipStream_t stream) {
  (void)in_sizes; (void)n_in; (void)out_size; (void)ws_size;
  const int B = 4;
  const size_t Pn = 2097152;  // B*128*4096

  float* Wf = (float*)d_ws;
  float* s_ = Wf;
  float* r_ = Wf + Pn;
  float* ctxp = Wf + 23 * Pn;
  float* ctxo = Wf + 24 * Pn;
  float* nrm  = Wf + 25 * Pn;  // nqw, nqz, nkw, nkz (16384 each)
  char* wb8 = (char*)d_ws + (25 * Pn + 65536) * sizeof(float);
#define WB(off) ((__bf16*)(wb8 + (size_t)(off) * 2))

  // --- 1. convert weights to bf16 ---
  static const struct { int idx; int off; int n; } cvt[15] = {
    {2, 0, 16384},      {3, 16384, 16384},  {4, 32768, 16384},  {5, 49152, 16384},
    {6, 65536, 16384},  {7, 81920, 16384},  {8, 98304, 16384},  {9, 114688, 16384},
    {10, 131072, 16384},{11, 147456, 16384},{12, 163840, 16384},{13, 180224, 16384},
    {14, 196608, 16384},{15, 212992, 16384},{16, 229376, 32768}};
  for (int i = 0; i < 15; ++i)
    k_cvt<<<(cvt[i].n + 255) / 256, 256, 0, stream>>>((const float*)d_in[cvt[i].idx],
                                                      WB(cvt[i].off), cvt[i].n);

  // --- 2. avgpool 4x4 ---
  k_avgpool<<<B * 128, 256, 0, stream>>>((const float*)d_in[0], s_);
  k_avgpool<<<B * 128, 256, 0, stream>>>((const float*)d_in[1], r_);

  // --- 3. evolves (q on s, k/v on r) ---
  const int offIn[3] = {0, 16384, 32768};
  const int offPw[3] = {49152, 65536, 81920};
  const int offOw[3] = {98304, 131072, 163840};
  const int offZ[3]  = {114688, 147456, 180224};
  dim3 gP(64, B);
  for (int e = 0; e < 3; ++e) {
    const float* inp = (e == 0) ? s_ : r_;
    float* fo = Wf + (2 + 7 * (size_t)e) * Pn;
    float* om = fo + Pn; float* ze = om + Pn;
    float* xa = ze + Pn; float* xb = xa + Pn;
    float* vv = xb + Pn; float* pw = vv + Pn;
    const float* wdw = (const float*)d_in[17 + e];
    // step 1 is analytic: x = F*DT^2, v = F*DT (fused into the force GEMM)
    k_conv1x1<3><<<gP, 256, 0, stream>>>(inp, WB(offIn[e]), fo, xa, vv);
    k_conv1x1<1><<<gP, 256, 0, stream>>>(inp, WB(offOw[e]), om, nullptr, nullptr);
    k_conv1x1<2><<<gP, 256, 0, stream>>>(inp, WB(offZ[e]),  ze, nullptr, nullptr);
    // steps 2, 3 (ping-pong x)
    k_conv1x1<0><<<gP, 256, 0, stream>>>(xa, WB(offPw[e]), pw, nullptr, nullptr);
    k_dwstep<<<B * 128, 256, 0, stream>>>(xa, pw, fo, om, ze, vv, xb, wdw);
    k_conv1x1<0><<<gP, 256, 0, stream>>>(xb, WB(offPw[e]), pw, nullptr, nullptr);
    k_dwstep<<<B * 128, 256, 0, stream>>>(xb, pw, fo, om, ze, vv, xa, wdw);
  }

  // --- 4. per-token squared norms of omega/zeta maps ---
  float* fo_q = Wf + 2 * Pn;  float* fo_k = Wf + 9 * Pn;  float* fo_v = Wf + 16 * Pn;
  k_norm<<<B * 16, 256, 0, stream>>>(fo_q + Pn,     nrm);          // |qw|^2
  k_norm<<<B * 16, 256, 0, stream>>>(fo_q + 2 * Pn, nrm + 16384);  // |qz|^2
  k_norm<<<B * 16, 256, 0, stream>>>(fo_k + Pn,     nrm + 32768);  // |kw|^2
  k_norm<<<B * 16, 256, 0, stream>>>(fo_k + 2 * Pn, nrm + 49152);  // |kz|^2

  // --- 5. flash attention ---
  k_flash<<<dim3(64, B), 128, 0, stream>>>(
      fo_q + 3 * Pn, fo_k + 3 * Pn, fo_v + 3 * Pn,   // q, k, v  (= x_a of each evolve)
      fo_q + Pn, fo_k + Pn, fo_q + 2 * Pn, fo_k + 2 * Pn,
      nrm, nrm + 16384, nrm + 32768, nrm + 49152, ctxp);

  // --- 6. Wout ---
  k_conv1x1<0><<<gP, 256, 0, stream>>>(ctxp, WB(196608), ctxo, nullptr, nullptr);

  // --- 7. fused upsample + gating + residual ---
  k_gating<<<dim3(1024, B), 256, 0, stream>>>((const float*)d_in[0], ctxo,
                                              WB(229376), WB(212992), (float*)d_out);
#undef WB
}
